// EagleAttention_32332513804773
// MI455X (gfx1250) — matrix operand order repaired and checked by reading, not compiled, not run on hardware
//
#include <hip/hip_runtime.h>
#include <hip/hip_bf16.h>

// ---------------------------------------------------------------------------
// EagleAttention for MI455X (gfx1250): bf16 WMMA GEMMs + flash attention.
// Pipeline: QKV proj -> RoPE -> causal GQA flash attention -> out proj.
// Register + LDS double-buffered staging: global loads for tile t+1 are in
// flight while WMMAs consume tile t. LDS layouts give each lane contiguous
// 16B-aligned fragment runs -> ds_load_b128.
// ---------------------------------------------------------------------------

typedef __attribute__((ext_vector_type(16))) __bf16 v16bf;
typedef __attribute__((ext_vector_type(8)))  float  v8f;

#define B_  2
#define S_  2048
#define H_  32
#define KV_ 8
#define D_  64
#define HID_ 2048

union BF4 { uint2 u; __bf16 h[4]; };

static __device__ __forceinline__ v8f wmma_bf16(v16bf a, v16bf b, v8f c) {
    return __builtin_amdgcn_wmma_f32_16x16x32_bf16(false, a, false, b, (short)0, c, false, false);
}

// A-fragment (16x32 bf16): lane r=lane&15, kh=lane>>4; one 32B run.
static __device__ __forceinline__ v16bf load_frag_a(const __bf16* base, int ld) {
    int lane = threadIdx.x & 31;
    int r = lane & 15, kh = lane >> 4;
    const __bf16* p = base + r * ld;
    v16bf a;
#pragma unroll
    for (int i = 0; i < 8; ++i) a[i]     = p[16 * kh + i];
#pragma unroll
    for (int i = 0; i < 8; ++i) a[8 + i] = p[16 * kh + 8 + i];
    return a;
}

// B-fragment (32x16 bf16) from transposed tile BsT[n][k]; one 32B run.
static __device__ __forceinline__ v16bf load_frag_bT(const __bf16* base, int ld) {
    int lane = threadIdx.x & 31;
    int n = lane & 15, kh = lane >> 4;
    const __bf16* p = base + n * ld + 16 * kh;
    v16bf b;
#pragma unroll
    for (int i = 0; i < 16; ++i) b[i] = p[i];
    return b;
}

// ---------------------------------------------------------------------------
// GEMM: C[M,N] = A[M,K] * W[K,N]  (fp32 in, bf16 WMMA, fp32 out)
// 256 threads = 8 waves; tile 128x128; wave = 32x64 patch (2x4 WMMA accs).
// Double-buffered LDS + register staging.
// ---------------------------------------------------------------------------
#define BM 128
#define BN 128
#define BK 32
#define LDA 40
#define LDB 40

__global__ __launch_bounds__(256) void gemm_bf16_wmma(
    const float* __restrict__ A, const float* __restrict__ W,
    float* __restrict__ C, int M, int N, int K) {
    __shared__ __bf16 As[2][BM][LDA];
    __shared__ __bf16 BsT[2][BN][LDB];   // BsT[n][k] = W[k0+k][n0+n]
    int tid  = threadIdx.x;
    int wave = tid >> 5;
    int waveM = wave & 3;
    int waveN = wave >> 2;
    int m0 = blockIdx.y * BM;
    int n0 = blockIdx.x * BN;

    // per-thread staging coordinates
    int arow[4], acol[4], brow[4], bcol[4];
#pragma unroll
    for (int t = 0; t < 4; ++t) {
        int idx = tid + t * 256;
        arow[t] = idx >> 3;  acol[t] = (idx & 7) * 4;    // A: 128x32
        brow[t] = idx >> 5;  bcol[t] = (idx & 31) * 4;   // W: 32x128
    }

    // preload first k-tile into registers (distinct regs -> clause + 1 wait)
    float4 areg[4], breg[4];
#pragma unroll
    for (int t = 0; t < 4; ++t) {
        areg[t] = *reinterpret_cast<const float4*>(A + (size_t)(m0 + arow[t]) * K + acol[t]);
        breg[t] = *reinterpret_cast<const float4*>(W + (size_t)brow[t] * N + n0 + bcol[t]);
    }

    v8f acc[2][4] = {};
    int cur = 0;
    for (int k0 = 0; k0 < K; k0 += BK) {
        // convert + store staged registers to LDS[cur]
#pragma unroll
        for (int t = 0; t < 4; ++t) {
            BF4 p;
            p.h[0] = (__bf16)areg[t].x; p.h[1] = (__bf16)areg[t].y;
            p.h[2] = (__bf16)areg[t].z; p.h[3] = (__bf16)areg[t].w;
            *reinterpret_cast<uint2*>(&As[cur][arow[t]][acol[t]]) = p.u;
            BsT[cur][bcol[t] + 0][brow[t]] = (__bf16)breg[t].x;
            BsT[cur][bcol[t] + 1][brow[t]] = (__bf16)breg[t].y;
            BsT[cur][bcol[t] + 2][brow[t]] = (__bf16)breg[t].z;
            BsT[cur][bcol[t] + 3][brow[t]] = (__bf16)breg[t].w;
        }
        // issue next tile's global loads; they overlap with compute below
        if (k0 + BK < K) {
#pragma unroll
            for (int t = 0; t < 4; ++t) {
                areg[t] = *reinterpret_cast<const float4*>(
                    A + (size_t)(m0 + arow[t]) * K + k0 + BK + acol[t]);
                breg[t] = *reinterpret_cast<const float4*>(
                    W + (size_t)(k0 + BK + brow[t]) * N + n0 + bcol[t]);
            }
        }
        __syncthreads();

        v16bf af[2];
#pragma unroll
        for (int mi = 0; mi < 2; ++mi)
            af[mi] = load_frag_a(&As[cur][waveM * 32 + mi * 16][0], LDA);
#pragma unroll
        for (int ni = 0; ni < 4; ++ni) {
            v16bf bf = load_frag_bT(&BsT[cur][waveN * 64 + ni * 16][0], LDB);
#pragma unroll
            for (int mi = 0; mi < 2; ++mi)
                acc[mi][ni] = wmma_bf16(af[mi], bf, acc[mi][ni]);
        }
        cur ^= 1;
    }

    int lane = tid & 31;
    int nlane = lane & 15, mh = lane >> 4;
#pragma unroll
    for (int mi = 0; mi < 2; ++mi)
#pragma unroll
        for (int ni = 0; ni < 4; ++ni) {
            float* Cp = C + (size_t)(m0 + waveM * 32 + mi * 16 + mh * 8) * N
                          + n0 + waveN * 64 + ni * 16 + nlane;
#pragma unroll
            for (int i = 0; i < 8; ++i) Cp[(size_t)i * N] = acc[mi][ni][i];
        }
}

// ---------------------------------------------------------------------------
// RoPE, in place on [B,S,nh,64]; position id == s (broadcast arange).
// ---------------------------------------------------------------------------
__global__ void rope_kernel(float* __restrict__ X, int nh, int total) {
    int idx = blockIdx.x * blockDim.x + threadIdx.x;
    if (idx >= total) return;
    int j   = idx & 31;
    int row = idx >> 5;
    int s   = (row / nh) % S_;
    float inv = exp2f(-(float)j * (13.287712379549449f / 32.0f)); // 10000^(-j/32)
    float ang = (float)s * inv;
    float c = __cosf(ang), sn = __sinf(ang);
    float* p = X + (size_t)row * 64;
    float x1 = p[j], x2 = p[j + 32];
    p[j]      = x1 * c - x2 * sn;
    p[j + 32] = x2 * c + x1 * sn;
}

// ---------------------------------------------------------------------------
// Flash attention, causal, GQA (rep=4). Block = (b, h, 128-q-row tile),
// 8 waves x 16 q-rows. Double-buffered KV tiles of 32 keys.
// ---------------------------------------------------------------------------
#define LDK 72   // D_ + 8
#define LDV 40   // 32 keys + 8
#define LDP 40

__global__ __launch_bounds__(256) void attn_kernel(
    const float* __restrict__ Q, const float* __restrict__ K,
    const float* __restrict__ V, float* __restrict__ AO) {
    int bh = blockIdx.x;
    int b  = bh >> 5;
    int h  = bh & 31;
    int hkv = h >> 2;
    int q0 = blockIdx.y * 128;
    int tid = threadIdx.x, wave = tid >> 5, lane = tid & 31;
    int qw = q0 + wave * 16;
    int nlane = lane & 15, mh = lane >> 4;

    __shared__ __bf16 Ks[2][32][LDK];    // [key][dim]
    __shared__ __bf16 VsT[2][D_][LDV];   // [dim][key]
    __shared__ __bf16 Ps[8][16][LDP];    // per-wave P patch

    // Q fragments (16x64 = two 16x32 A-frags), vectorized fp32 loads
    v16bf qf[2];
    {
        const float* Qrow = Q + (size_t)(b * S_ + qw + nlane) * (H_ * D_) + h * D_;
#pragma unroll
        for (int kk = 0; kk < 2; ++kk) {
            float4 u0 = *reinterpret_cast<const float4*>(Qrow + kk * 32 + 16 * mh);
            float4 u1 = *reinterpret_cast<const float4*>(Qrow + kk * 32 + 16 * mh + 4);
            float4 u2 = *reinterpret_cast<const float4*>(Qrow + kk * 32 + 16 * mh + 8);
            float4 u3 = *reinterpret_cast<const float4*>(Qrow + kk * 32 + 16 * mh + 12);
            v16bf a;
            a[0]  = (__bf16)u0.x; a[1]  = (__bf16)u0.y; a[2]  = (__bf16)u0.z; a[3]  = (__bf16)u0.w;
            a[4]  = (__bf16)u1.x; a[5]  = (__bf16)u1.y; a[6]  = (__bf16)u1.z; a[7]  = (__bf16)u1.w;
            a[8]  = (__bf16)u2.x; a[9]  = (__bf16)u2.y; a[10] = (__bf16)u2.z; a[11] = (__bf16)u2.w;
            a[12] = (__bf16)u3.x; a[13] = (__bf16)u3.y; a[14] = (__bf16)u3.z; a[15] = (__bf16)u3.w;
            qf[kk] = a;
        }
    }

    // staging coordinates + preload first KV tile
    int krow[2], kcol[2];
#pragma unroll
    for (int t = 0; t < 2; ++t) {
        int idx = tid + t * 256;
        krow[t] = idx >> 4;  kcol[t] = (idx & 15) * 4;
    }
    float4 kreg[2], vreg[2];
#pragma unroll
    for (int t = 0; t < 2; ++t) {
        size_t g = (size_t)(b * S_ + krow[t]) * (KV_ * D_) + hkv * D_ + kcol[t];
        kreg[t] = *reinterpret_cast<const float4*>(K + g);
        vreg[t] = *reinterpret_cast<const float4*>(V + g);
    }

    v8f o_acc[4] = {};
    float mrow[8], lrow[8];
#pragma unroll
    for (int i = 0; i < 8; ++i) { mrow[i] = -1e30f; lrow[i] = 0.0f; }

    int cur = 0;
    int kv_end = q0 + 128;
    for (int kv0 = 0; kv0 < kv_end; kv0 += 32) {
        // store staged K/V into LDS[cur]
#pragma unroll
        for (int t = 0; t < 2; ++t) {
            BF4 p;
            p.h[0] = (__bf16)kreg[t].x; p.h[1] = (__bf16)kreg[t].y;
            p.h[2] = (__bf16)kreg[t].z; p.h[3] = (__bf16)kreg[t].w;
            *reinterpret_cast<uint2*>(&Ks[cur][krow[t]][kcol[t]]) = p.u;
            VsT[cur][kcol[t] + 0][krow[t]] = (__bf16)vreg[t].x;
            VsT[cur][kcol[t] + 1][krow[t]] = (__bf16)vreg[t].y;
            VsT[cur][kcol[t] + 2][krow[t]] = (__bf16)vreg[t].z;
            VsT[cur][kcol[t] + 3][krow[t]] = (__bf16)vreg[t].w;
        }
        // issue next tile's global loads (overlap with compute)
        if (kv0 + 32 < kv_end) {
#pragma unroll
            for (int t = 0; t < 2; ++t) {
                size_t g = (size_t)(b * S_ + kv0 + 32 + krow[t]) * (KV_ * D_) + hkv * D_ + kcol[t];
                kreg[t] = *reinterpret_cast<const float4*>(K + g);
                vreg[t] = *reinterpret_cast<const float4*>(V + g);
            }
        }
        __syncthreads();

        // S = Q * K^T : two 16-key groups, k-dim 64 in 2 WMMA steps
        v8f s_t[2];
#pragma unroll
        for (int nk = 0; nk < 2; ++nk) {
            v8f acc = {};
#pragma unroll
            for (int kk = 0; kk < 2; ++kk) {
                const __bf16* p = &Ks[cur][nk * 16 + nlane][kk * 32 + 16 * mh];
                v16bf bf;
#pragma unroll
                for (int i = 0; i < 16; ++i) bf[i] = p[i];
                acc = wmma_bf16(qf[kk], bf, acc);
            }
            s_t[nk] = acc;
        }

        // scale + causal mask + online softmax
        float newmax[8];
#pragma unroll
        for (int i = 0; i < 8; ++i) {
            int qrow = qw + mh * 8 + i;
            float lm = -1e30f;
#pragma unroll
            for (int nk = 0; nk < 2; ++nk) {
                int key = kv0 + nk * 16 + nlane;
                float sv = s_t[nk][i] * 0.125f;
                sv = (key <= qrow) ? sv : -1e30f;
                s_t[nk][i] = sv;
                lm = fmaxf(lm, sv);
            }
#pragma unroll
            for (int off = 1; off < 16; off <<= 1)
                lm = fmaxf(lm, __shfl_xor(lm, off, 32));
            newmax[i] = fmaxf(mrow[i], lm);
        }
#pragma unroll
        for (int i = 0; i < 8; ++i) {
            float alpha = __expf(mrow[i] - newmax[i]);
            float rs = 0.0f;
#pragma unroll
            for (int nk = 0; nk < 2; ++nk) {
                float p = __expf(s_t[nk][i] - newmax[i]);
                s_t[nk][i] = p;
                rs += p;
            }
#pragma unroll
            for (int off = 1; off < 16; off <<= 1)
                rs += __shfl_xor(rs, off, 32);
            lrow[i] = lrow[i] * alpha + rs;
            mrow[i] = newmax[i];
#pragma unroll
            for (int j = 0; j < 4; ++j) o_acc[j][i] *= alpha;
        }

        // C-layout -> A-layout via per-wave LDS patch
#pragma unroll
        for (int nk = 0; nk < 2; ++nk)
#pragma unroll
            for (int i = 0; i < 8; ++i)
                Ps[wave][mh * 8 + i][nk * 16 + nlane] = (__bf16)s_t[nk][i];
        __syncthreads();
        v16bf pf = load_frag_a(&Ps[wave][0][0], LDP);

        // O += P * V
#pragma unroll
        for (int j = 0; j < 4; ++j) {
            const __bf16* p = &VsT[cur][j * 16 + nlane][16 * mh];
            v16bf vf;
#pragma unroll
            for (int i = 0; i < 16; ++i) vf[i] = p[i];
            o_acc[j] = wmma_bf16(pf, vf, o_acc[j]);
        }
        cur ^= 1;
    }

    // normalize and store fp32 [B,S,H*D]
#pragma unroll
    for (int j = 0; j < 4; ++j) {
        float* Op = AO + (size_t)(b * S_ + qw + mh * 8) * (H_ * D_) + h * D_ + j * 16 + nlane;
#pragma unroll
        for (int i = 0; i < 8; ++i)
            Op[(size_t)i * (H_ * D_)] = o_acc[j][i] / lrow[i];
    }
}

// ---------------------------------------------------------------------------
extern "C" void kernel_launch(void* const* d_in, const int* in_sizes, int n_in,
                              void* d_out, int out_size, void* d_ws, size_t ws_size,
                              hipStream_t stream) {
    const float* X  = (const float*)d_in[0];   // [B,S,2*HID] = [4096,4096]
    const float* Wq = (const float*)d_in[1];   // [4096,2048]
    const float* Wk = (const float*)d_in[2];   // [4096,512]
    const float* Wv = (const float*)d_in[3];   // [4096,512]
    const float* Wo = (const float*)d_in[4];   // [2048,2048]
    (void)in_sizes; (void)n_in; (void)out_size; (void)ws_size;

    const int M = B_ * S_;       // 4096
    const int Kx = 2 * HID_;     // 4096
    float* Qb  = (float*)d_ws;                       // 4096x2048
    float* Kb  = Qb + (size_t)M * (H_ * D_);         // 4096x512
    float* Vb  = Kb + (size_t)M * (KV_ * D_);        // 4096x512
    float* AOb = Vb + (size_t)M * (KV_ * D_);        // 4096x2048

    gemm_bf16_wmma<<<dim3((H_ * D_) / BN, M / BM), 256, 0, stream>>>(X, Wq, Qb, M, H_ * D_, Kx);
    gemm_bf16_wmma<<<dim3((KV_ * D_) / BN, M / BM), 256, 0, stream>>>(X, Wk, Kb, M, KV_ * D_, Kx);
    gemm_bf16_wmma<<<dim3((KV_ * D_) / BN, M / BM), 256, 0, stream>>>(X, Wv, Vb, M, KV_ * D_, Kx);

    int totq = M * H_ * 32, totk = M * KV_ * 32;
    rope_kernel<<<(totq + 255) / 256, 256, 0, stream>>>(Qb, H_, totq);
    rope_kernel<<<(totk + 255) / 256, 256, 0, stream>>>(Kb, KV_, totk);

    attn_kernel<<<dim3(B_ * H_, S_ / 128), 256, 0, stream>>>(Qb, Kb, Vb, AOb);

    gemm_bf16_wmma<<<dim3(HID_ / BN, M / BM), 256, 0, stream>>>(AOb, Wo, (float*)d_out, M, HID_, H_ * D_);
}
